// Kinematics_LSTM_decoder_21534966022553
// MI455X (gfx1250) — matrix operand order repaired and checked by reading, not compiled, not run on hardware
//
#include <hip/hip_runtime.h>
#include <cstdint>
#include <cstddef>

// ---------------------------------------------------------------------------
// Types for CDNA5 WMMA (gfx1250, wave32)
// ---------------------------------------------------------------------------
typedef __bf16 bf16;
typedef __attribute__((ext_vector_type(16))) __bf16 v16bf;
typedef __attribute__((ext_vector_type(8)))  __bf16 v8bf;
typedef __attribute__((ext_vector_type(8)))  float   v8f;

#define DEV __device__ __forceinline__

static constexpr int B    = 64;
static constexpr int H    = 1152;
static constexpr int G4   = 4 * H;     // 4608 gate rows
static constexpr int TOUT = 25;
static constexpr int TENC = 49;
static constexpr int DIN  = 54;
static constexpr int KX0  = 64;        // padded K for cell-0 input GEMM

DEV float sigm(float x)      { return 1.0f / (1.0f + __expf(-x)); }
DEV float tanh_fast(float x) { return 2.0f / (1.0f + __expf(-2.0f * x)) - 1.0f; }

// A-fragment (16x32 bf16, M=lane%16): lanes<16 hold K=k0+0..7 & k0+16..23,
// lanes>=16 hold K=k0+8..15 & k0+24..31  (ISA 7.12.2)
DEV v16bf load_a(const bf16* __restrict__ row, int k0, int half) {
  v8bf lo = *(const v8bf*)(row + k0 + half * 8);
  v8bf hi = *(const v8bf*)(row + k0 + 16 + half * 8);
  return __builtin_shufflevector(lo, hi, 0,1,2,3,4,5,6,7,8,9,10,11,12,13,14,15);
}

DEV v8f wmma_bf16(v16bf a, v16bf b, v8f c) {
  // (neg_a, A, neg_b, B, c_mod, C, reuse_a, reuse_b)
  return __builtin_amdgcn_wmma_f32_16x16x32_bf16(false, a, false, b, (short)0, c,
                                                 false, false);
}

// ---------------------------------------------------------------------------
// One LSTM cell stage:  g = X @ Wx^T + Hprev @ Wh^T (+ biases), gate fuse.
// grid = H/16 blocks (16 hidden columns each), 128 threads = 4 waves.
// Wave w owns batch rows [16w,16w+16); accumulates 4 gate tiles (i,f,g,o).
// ---------------------------------------------------------------------------
__global__ __launch_bounds__(128) void lstm_stage_kernel(
    const bf16*  __restrict__ X,  int Kx,              // [B, Kx] bf16
    const bf16*  __restrict__ Hp,                      // [B, H]  bf16 (recurrent)
    const bf16*  __restrict__ Wx,                      // [4H, Kx] bf16
    const bf16*  __restrict__ Wh,                      // [4H, H]  bf16
    const float* __restrict__ bih, const float* __restrict__ bhh,
    const float* __restrict__ c_old,
    float* __restrict__ c_new,
    float* __restrict__ h_new_f,
    bf16*  __restrict__ h_new_b)
{
  const int lane = threadIdx.x & 31;
  const int wave = threadIdx.x >> 5;
  const int half = lane >> 4;        // K-half select for fragments
  const int lm   = lane & 15;        // A: row M, B: col N, D: col N
  const int j0   = blockIdx.x * 16;  // hidden-column tile
  const int mA   = wave * 16 + lm;   // batch row this lane loads for A

  v8f acc0 = {0,0,0,0,0,0,0,0};
  v8f acc1 = acc0, acc2 = acc0, acc3 = acc0;

  // ---- input GEMM:  X @ Wx^T ----
  {
    const bf16*  xr      = X  + (size_t)mA * Kx;
    const bf16*  w0      = Wx + (size_t)(j0 + lm) * Kx;
    const size_t gstride = (size_t)H * Kx;             // stride between gates
    for (int k0 = 0; k0 < Kx; k0 += 32) {
      v16bf a = load_a(xr, k0, half);
      const bf16* w = w0;
      v16bf b0 = load_a(w, k0, half);
      v16bf b1 = load_a(w + gstride, k0, half);
      v16bf b2 = load_a(w + 2 * gstride, k0, half);
      v16bf b3 = load_a(w + 3 * gstride, k0, half);
      acc0 = wmma_bf16(a, b0, acc0);
      acc1 = wmma_bf16(a, b1, acc1);
      acc2 = wmma_bf16(a, b2, acc2);
      acc3 = wmma_bf16(a, b3, acc3);
    }
  }
  // ---- recurrent GEMM:  Hprev @ Wh^T  (K = H) ----
  {
    const bf16*  hr      = Hp + (size_t)mA * H;
    const bf16*  w0      = Wh + (size_t)(j0 + lm) * H;
    const size_t gstride = (size_t)H * H;
    for (int k0 = 0; k0 < H; k0 += 32) {
      v16bf a = load_a(hr, k0, half);
      const bf16* w = w0;
      v16bf b0 = load_a(w, k0, half);
      v16bf b1 = load_a(w + gstride, k0, half);
      v16bf b2 = load_a(w + 2 * gstride, k0, half);
      v16bf b3 = load_a(w + 3 * gstride, k0, half);
      acc0 = wmma_bf16(a, b0, acc0);
      acc1 = wmma_bf16(a, b1, acc1);
      acc2 = wmma_bf16(a, b2, acc2);
      acc3 = wmma_bf16(a, b3, acc3);
    }
  }

  // ---- fused gates. D layout: lane holds col N=lm; VGPR v -> row v+8*half.
  const int j = j0 + lm;
  const float bs0 = bih[j]          + bhh[j];
  const float bs1 = bih[H   + j]    + bhh[H   + j];
  const float bs2 = bih[2*H + j]    + bhh[2*H + j];
  const float bs3 = bih[3*H + j]    + bhh[3*H + j];
#pragma unroll
  for (int v = 0; v < 8; ++v) {
    const int m = wave * 16 + half * 8 + v;
    const size_t idx = (size_t)m * H + j;
    const float iv = sigm(acc0[v] + bs0);
    const float fv = sigm(acc1[v] + bs1);
    const float gv = tanh_fast(acc2[v] + bs2);
    const float ov = sigm(acc3[v] + bs3);
    const float cn = fv * c_old[idx] + iv * gv;
    const float hn = ov * tanh_fast(cn);
    c_new[idx]   = cn;
    h_new_f[idx] = hn;
    h_new_b[idx] = (bf16)hn;
  }
}

// ---------------------------------------------------------------------------
// Output heads + next-step x0 (padded to 64).  grid = B blocks x 64 threads.
// pre[j] = h_sel . W_sel[:, col] + b_sel[col] + x0_cur[j]
// ---------------------------------------------------------------------------
__global__ __launch_bounds__(64) void head_kernel(
    const float* __restrict__ h1, const float* __restrict__ h2,
    const float* __restrict__ h3, const float* __restrict__ h4,
    const float* __restrict__ h5,
    const float* __restrict__ Wsp, const float* __restrict__ bsp,
    const float* __restrict__ Wl1, const float* __restrict__ bl1,
    const float* __restrict__ Wl2, const float* __restrict__ bl2,
    const float* __restrict__ Wa1, const float* __restrict__ ba1,
    const float* __restrict__ Wa2, const float* __restrict__ ba2,
    const float* __restrict__ x0_cur,
    float* __restrict__ x0_next_f, bf16* __restrict__ x0_next_b,
    float* __restrict__ out, int t)
{
  const int b = blockIdx.x;
  const int j = threadIdx.x;          // 0..63
  float val = 0.0f;
  if (j < DIN) {
    const float* h; const float* W; const float* bb; int col; int dim;
    if      (j < 12) { h = h2; W = Wl1; bb = bl1; col = j;      dim = 12; }
    else if (j < 24) { h = h3; W = Wl2; bb = bl2; col = j - 12; dim = 12; }
    else if (j < 36) { h = h1; W = Wsp; bb = bsp; col = j - 24; dim = 12; }
    else if (j < 45) { h = h4; W = Wa1; bb = ba1; col = j - 36; dim = 9;  }
    else             { h = h5; W = Wa2; bb = ba2; col = j - 45; dim = 9;  }
    const float* hb = h + (size_t)b * H;
    float s = 0.0f;
    for (int k = 0; k < H; ++k) s += hb[k] * W[k * dim + col];
    val = s + bb[col] + x0_cur[b * KX0 + j];
    out[((size_t)b * TOUT + t) * DIN + j] = val;
  }
  x0_next_f[b * KX0 + j] = val;        // 0 in pad lanes
  x0_next_b[b * KX0 + j] = (bf16)val;
}

// ---------------------------------------------------------------------------
// Initialization kernels
// ---------------------------------------------------------------------------
__global__ void zero_u32_kernel(uint32_t* p, size_t n) {
  size_t i = (size_t)blockIdx.x * blockDim.x + threadIdx.x;
  const size_t stride = (size_t)gridDim.x * blockDim.x;
  for (; i < n; i += stride) p[i] = 0u;
}

__global__ void convert_pad_kernel(const float* __restrict__ src,
                                   bf16* __restrict__ dst,
                                   int rows, int scols, int dcols) {
  const size_t n = (size_t)rows * dcols;
  size_t i = (size_t)blockIdx.x * blockDim.x + threadIdx.x;
  const size_t stride = (size_t)gridDim.x * blockDim.x;
  for (; i < n; i += stride) {
    const int r = (int)(i / dcols);
    const int c = (int)(i % dcols);
    dst[i] = (c < scols) ? (bf16)src[(size_t)r * scols + c] : (bf16)0.0f;
  }
}

__global__ void init_states_kernel(const float* __restrict__ hs,
                                   const float* __restrict__ cs,
                                   const float* __restrict__ gt,
                                   float* __restrict__ h0f, bf16* __restrict__ h0b,
                                   float* __restrict__ c0,
                                   float* __restrict__ h1f, bf16* __restrict__ h1b,
                                   float* __restrict__ c1) {
  const int idx = blockIdx.x * blockDim.x + threadIdx.x;
  if (idx >= B * H) return;
  const int b = idx / H, j = idx % H;
  const float* hp = hs + (size_t)b * TENC * H + j;
  const float* cp = cs + (size_t)b * TENC * H + j;
  float sh = 0.0f, sc = 0.0f;
  for (int t = 0; t < TENC; ++t) { sh += hp[t * H]; sc += cp[t * H]; }
  const float h0 = sh / (float)TENC;
  const float cc = sc / (float)TENC;
  const float h1 = (gt[(size_t)b * H + j] + sh) / (float)(TENC + 1);
  h0f[idx] = h0; h0b[idx] = (bf16)h0; c0[idx] = cc;
  h1f[idx] = h1; h1b[idx] = (bf16)h1; c1[idx] = cc;
}

__global__ void init_x0_kernel(const float* __restrict__ p,
                               float* __restrict__ x0f, bf16* __restrict__ x0b) {
  const int idx = blockIdx.x * blockDim.x + threadIdx.x;
  if (idx >= B * KX0) return;
  const int b = idx >> 6, k = idx & 63;
  const float v = (k < DIN) ? p[b * DIN + k] : 0.0f;
  x0f[idx] = v; x0b[idx] = (bf16)v;
}

// ---------------------------------------------------------------------------
// Host orchestration
// ---------------------------------------------------------------------------
extern "C" void kernel_launch(void* const* d_in, const int* in_sizes, int n_in,
                              void* d_out, int out_size, void* d_ws, size_t ws_size,
                              hipStream_t stream) {
  (void)in_sizes; (void)n_in; (void)out_size; (void)ws_size;

  // ---- inputs (setup_inputs order) ----
  const float* hidden = (const float*)d_in[0];
  const float* cells  = (const float*)d_in[1];
  const float* gstate = (const float*)d_in[2];
  const float* pin    = (const float*)d_in[3];
  const float* Wih[4]; const float* Whh[4]; const float* bih[4]; const float* bhh[4];
  for (int i = 0; i < 4; ++i) {
    Wih[i] = (const float*)d_in[4 + 4 * i + 0];
    Whh[i] = (const float*)d_in[4 + 4 * i + 1];
    bih[i] = (const float*)d_in[4 + 4 * i + 2];
    bhh[i] = (const float*)d_in[4 + 4 * i + 3];
  }
  const float* Wsp = (const float*)d_in[20]; const float* bsp = (const float*)d_in[21];
  const float* Wl1 = (const float*)d_in[22]; const float* bl1 = (const float*)d_in[23];
  const float* Wl2 = (const float*)d_in[24]; const float* bl2 = (const float*)d_in[25];
  const float* Wa1 = (const float*)d_in[26]; const float* ba1 = (const float*)d_in[27];
  const float* Wa2 = (const float*)d_in[28]; const float* ba2 = (const float*)d_in[29];
  float* out = (float*)d_out;

  // ---- workspace carve-out ----
  char* wp = (char*)d_ws;
  auto alloc = [&](size_t bytes) -> char* {
    char* r = wp;
    wp += (bytes + 255) & ~(size_t)255;
    return r;
  };

  // bf16 weights (L2-resident: ~75 MB)
  bf16* Wx0 = (bf16*)alloc((size_t)G4 * KX0 * sizeof(bf16));     // padded Wih0
  bf16* Wxw[4]; bf16* Whw[4];
  Wxw[0] = Wx0;
  Whw[0] = (bf16*)alloc((size_t)G4 * H * sizeof(bf16));
  for (int i = 1; i < 4; ++i) {
    Wxw[i] = (bf16*)alloc((size_t)G4 * H * sizeof(bf16));
    Whw[i] = (bf16*)alloc((size_t)G4 * H * sizeof(bf16));
  }

  // double-buffered states (parity = step & 1)
  char* state_base = wp;
  float* hf[6][2]; bf16* hb[6][2]; float* cst[6][2];
  for (int i = 0; i < 6; ++i)
    for (int s = 0; s < 2; ++s) hf[i][s] = (float*)alloc((size_t)B * H * sizeof(float));
  for (int i = 0; i < 6; ++i)
    for (int s = 0; s < 2; ++s) hb[i][s] = (bf16*)alloc((size_t)B * H * sizeof(bf16));
  for (int i = 0; i < 6; ++i)
    for (int s = 0; s < 2; ++s) cst[i][s] = (float*)alloc((size_t)B * H * sizeof(float));
  float* x0f[2]; bf16* x0b[2];
  for (int s = 0; s < 2; ++s) x0f[s] = (float*)alloc((size_t)B * KX0 * sizeof(float));
  for (int s = 0; s < 2; ++s) x0b[s] = (bf16*)alloc((size_t)B * KX0 * sizeof(bf16));
  const size_t state_bytes = (size_t)(wp - state_base);

  // ---- weight conversion (fp32 -> bf16, pad Wih0 to K=64) ----
  convert_pad_kernel<<<1024, 256, 0, stream>>>(Wih[0], Wx0, G4, DIN, KX0);
  convert_pad_kernel<<<4096, 256, 0, stream>>>(Whh[0], Whw[0], G4, H, H);
  for (int i = 1; i < 4; ++i) {
    convert_pad_kernel<<<4096, 256, 0, stream>>>(Wih[i], Wxw[i], G4, H, H);
    convert_pad_kernel<<<4096, 256, 0, stream>>>(Whh[i], Whw[i], G4, H, H);
  }

  // ---- state init ----
  zero_u32_kernel<<<2048, 256, 0, stream>>>((uint32_t*)state_base, state_bytes / 4);
  init_states_kernel<<<(B * H + 255) / 256, 256, 0, stream>>>(
      hidden, cells, gstate,
      hf[0][0], hb[0][0], cst[0][0],
      hf[1][0], hb[1][0], cst[1][0]);
  init_x0_kernel<<<(B * KX0 + 255) / 256, 256, 0, stream>>>(pin, x0f[0], x0b[0]);

  // ---- 25 time steps; 6 LSTM stages + head per step ----
  const int nblk = H / 16;  // 72
  for (int t = 0; t < TOUT; ++t) {
    const int cur = t & 1, nxt = cur ^ 1;
    // cell 0: x = padded x0
    lstm_stage_kernel<<<nblk, 128, 0, stream>>>(
        x0b[cur], KX0, hb[0][cur], Wxw[0], Whw[0], bih[0], bhh[0],
        cst[0][cur], cst[0][nxt], hf[0][nxt], hb[0][nxt]);
    // cell 1: x = h0_new
    lstm_stage_kernel<<<nblk, 128, 0, stream>>>(
        hb[0][nxt], H, hb[1][cur], Wxw[1], Whw[1], bih[1], bhh[1],
        cst[1][cur], cst[1][nxt], hf[1][nxt], hb[1][nxt]);
    // cells 2 & 3: x = h1_new, weights A
    lstm_stage_kernel<<<nblk, 128, 0, stream>>>(
        hb[1][nxt], H, hb[2][cur], Wxw[2], Whw[2], bih[2], bhh[2],
        cst[2][cur], cst[2][nxt], hf[2][nxt], hb[2][nxt]);
    lstm_stage_kernel<<<nblk, 128, 0, stream>>>(
        hb[1][nxt], H, hb[3][cur], Wxw[2], Whw[2], bih[2], bhh[2],
        cst[3][cur], cst[3][nxt], hf[3][nxt], hb[3][nxt]);
    // cell 4: x = h3_new, weights L
    lstm_stage_kernel<<<nblk, 128, 0, stream>>>(
        hb[3][nxt], H, hb[4][cur], Wxw[3], Whw[3], bih[3], bhh[3],
        cst[4][cur], cst[4][nxt], hf[4][nxt], hb[4][nxt]);
    // cell 5: x = h4_new
    lstm_stage_kernel<<<nblk, 128, 0, stream>>>(
        hb[4][nxt], H, hb[5][cur], Wxw[3], Whw[3], bih[3], bhh[3],
        cst[5][cur], cst[5][nxt], hf[5][nxt], hb[5][nxt]);
    // heads + next x0
    head_kernel<<<B, 64, 0, stream>>>(
        hf[1][nxt], hf[2][nxt], hf[3][nxt], hf[4][nxt], hf[5][nxt],
        Wsp, bsp, Wl1, bl1, Wl2, bl2, Wa1, ba1, Wa2, ba2,
        x0f[cur], x0f[nxt], x0b[nxt], out, t);
  }
}
